// SelfAttentionPooling_47021301957262
// MI455X (gfx1250) — matrix operand order repaired and checked by reading, not compiled, not run on hardware
//
#include <hip/hip_runtime.h>
#include <math.h>

// Problem constants (from reference)
#define B_  8192
#define S_  200
#define SP_ 208      // S padded to 13 tiles of 16
#define D_  64
#define H_  64

typedef __attribute__((ext_vector_type(16))) __bf16 v16bf;
typedef __attribute__((ext_vector_type(8)))  __bf16 v8bf;
typedef __attribute__((ext_vector_type(4)))  __bf16 v4bf;
typedef __attribute__((ext_vector_type(8)))  float  v8f;

__device__ __forceinline__ float fast_tanh(float x) {
#if __has_builtin(__builtin_amdgcn_tanhf)
  return __builtin_amdgcn_tanhf(x);   // hardware v_tanh_f32 (TRANS32, co-executes with WMMA)
#else
  return tanhf(x);
#endif
}

struct __align__(16) Smem {
  float  embF[S_ * D_];    // fp32 embeddings (pooling path)
  __bf16 embH[SP_ * D_];   // bf16 embeddings (WMMA A)
  __bf16 w1t[H_ * D_];     // W1^T bf16, [h][d] (WMMA B: contiguous K per column)
  float  scores[SP_];
  float  attn[SP_];
  float  red[256];
  float  part[256];
  int    sIds[SP_];
};

__global__ __launch_bounds__(256)
void attn_pool_kernel(const float* __restrict__ emb_table,
                      const float* __restrict__ W1,
                      const float* __restrict__ b1,
                      const float* __restrict__ W2,
                      const float* __restrict__ b2,
                      const int*   __restrict__ ids,
                      const int*   __restrict__ lengths,
                      float*       __restrict__ out)
{
  extern __shared__ char smem_raw[];
  Smem& sm = *reinterpret_cast<Smem*>(smem_raw);

  const int t    = threadIdx.x;
  const int lane = t & 31;
  const int wave = t >> 5;
  const int b    = blockIdx.x;

  // ---------------- Phase 0: stage ids and W1^T (bf16) ----------------
  for (int s = t; s < SP_; s += 256)
    sm.sIds[s] = (s < S_) ? ids[(size_t)b * S_ + s] : 0;
  for (int e = t; e < D_ * H_; e += 256) {
    int d = e >> 6, h = e & 63;            // W1 is [D][H] row-major
    sm.w1t[h * D_ + d] = (__bf16)W1[e];
  }
  __syncthreads();

  // ---------------- Phase 1: gather embeddings (fp32 + bf16) ----------
  {
    const int q  = t & 15;                 // float4 slot within a 64-wide row
    const int sg = t >> 4;                 // 16 rows in flight
    for (int s = sg; s < SP_; s += 16) {
      float4 v = make_float4(0.f, 0.f, 0.f, 0.f);
      if (s < S_) {
        v = reinterpret_cast<const float4*>(emb_table)[(size_t)sm.sIds[s] * (D_/4) + q];
        reinterpret_cast<float4*>(sm.embF)[s * (D_/4) + q] = v;
      }
      v4bf hv = { (__bf16)v.x, (__bf16)v.y, (__bf16)v.z, (__bf16)v.w };
      *reinterpret_cast<v4bf*>(&sm.embH[s * D_ + q * 4]) = hv;
    }
  }
  __syncthreads();

  // Per-lane constants for the score epilogue (C layout: N = lane%16)
  const int nidx = lane & 15;
  float b1v[4], w2v[4];
  #pragma unroll
  for (int nt = 0; nt < 4; ++nt) {
    b1v[nt] = b1[nt * 16 + nidx];
    w2v[nt] = W2[nt * 16 + nidx];
  }
  const float b2v = b2[0];

  // Preload all B fragments: 4 N-tiles x 2 K-steps.
  // B 32x16 bf16, this code's contiguous K order (same as A): lanes 0-15 col N, K=k0..k0+15; lanes 16-31 col N, K=k0+16..k0+31.
  v16bf Bf[4][2];
  #pragma unroll
  for (int nt = 0; nt < 4; ++nt) {
    #pragma unroll
    for (int ks = 0; ks < 2; ++ks) {
      const int n  = nt * 16 + nidx;
      const int k0 = ks * 32 + ((lane < 16) ? 0 : 16);
      const v8bf* wp = reinterpret_cast<const v8bf*>(&sm.w1t[n * D_ + k0]);
      v8bf lo = wp[0], hi = wp[1];
      #pragma unroll
      for (int j = 0; j < 8; ++j) { Bf[nt][ks][j] = lo[j]; Bf[nt][ks][j + 8] = hi[j]; }
    }
  }

  // ---------------- Phase 2: scores = tanh(emb@W1 + b1) @ W2 + b2 ------
  // Each wave owns M-tiles mt = wave, wave+8, ... (13 tiles of 16 tokens).
  for (int mt = wave; mt < SP_ / 16; mt += 8) {
    const int row = mt * 16 + nidx;        // A: M = lane%16 for both halves
    v16bf Af[2];
    #pragma unroll
    for (int ks = 0; ks < 2; ++ks) {
      // A 16x32 bf16, this code's contiguous K order (same as B): lane<16 -> K 0..15 +kb; lane>=16 -> K 16..31 +kb
      const int k0 = ks * 32 + ((lane < 16) ? 0 : 16);
      v8bf lo = *reinterpret_cast<const v8bf*>(&sm.embH[row * D_ + k0]);
      v8bf hi = *reinterpret_cast<const v8bf*>(&sm.embH[row * D_ + k0 + 8]);
      #pragma unroll
      for (int j = 0; j < 8; ++j) { Af[ks][j] = lo[j]; Af[ks][j + 8] = hi[j]; }
    }

    // Issue all 8 WMMAs back-to-back into 4 accumulators (pipelined XDL),
    // then run the tanh/W2 epilogue once.
    v8f cacc[4];
    #pragma unroll
    for (int nt = 0; nt < 4; ++nt) {
      v8f c = {0.f, 0.f, 0.f, 0.f, 0.f, 0.f, 0.f, 0.f};
      c = __builtin_amdgcn_wmma_f32_16x16x32_bf16(false, Af[0], false, Bf[nt][0],
                                                  (short)0, c, false, false);
      c = __builtin_amdgcn_wmma_f32_16x16x32_bf16(false, Af[1], false, Bf[nt][1],
                                                  (short)0, c, false, false);
      cacc[nt] = c;
    }

    float p[8];
    #pragma unroll
    for (int i = 0; i < 8; ++i) p[i] = 0.f;
    #pragma unroll
    for (int nt = 0; nt < 4; ++nt) {
      #pragma unroll
      for (int i = 0; i < 8; ++i)
        p[i] += fast_tanh(cacc[nt][i] + b1v[nt]) * w2v[nt];
    }

    // Reduce over the 16 N-lanes (masks 1..8 stay within each half-wave)
    #pragma unroll
    for (int m = 1; m < 16; m <<= 1) {
      #pragma unroll
      for (int i = 0; i < 8; ++i) p[i] += __shfl_xor(p[i], m, 32);
    }
    if (nidx == 0) {
      const int base = mt * 16 + ((lane >> 4) << 3);  // lane0 -> M 0..7, lane16 -> M 8..15
      #pragma unroll
      for (int i = 0; i < 8; ++i) sm.scores[base + i] = p[i] + b2v;
    }
  }
  __syncthreads();

  // ---------------- Phase 3: masked softmax over S ----------------------
  const int len = lengths[b];              // >= 1 by construction
  float mx = -__builtin_inff();
  for (int s = t; s < len; s += 256) mx = fmaxf(mx, sm.scores[s]);
  sm.red[t] = mx; __syncthreads();
  for (int o = 128; o > 0; o >>= 1) {
    if (t < o) sm.red[t] = fmaxf(sm.red[t], sm.red[t + o]);
    __syncthreads();
  }
  mx = sm.red[0]; __syncthreads();

  float sum = 0.f;
  for (int s = t; s < S_; s += 256) {
    float a = 0.f;
    if (s < len) { a = __expf(sm.scores[s] - mx); sum += a; }
    sm.attn[s] = a;
  }
  sm.red[t] = sum; __syncthreads();
  for (int o = 128; o > 0; o >>= 1) {
    if (t < o) sm.red[t] += sm.red[t + o];
    __syncthreads();
  }
  const float inv = 1.0f / sm.red[0];

  // ---------------- Phase 4: weighted-sum pooling (fp32) ----------------
  const int d = t & 63, g = t >> 6;
  float acc = 0.f;
  for (int s = g; s < S_; s += 4) acc += sm.attn[s] * sm.embF[s * D_ + d];
  sm.part[t] = acc; __syncthreads();
  if (t < 64) {
    float r = (sm.part[t] + sm.part[64 + t] + sm.part[128 + t] + sm.part[192 + t]) * inv;
    out[(size_t)b * D_ + t] = r;
  }
}

extern "C" void kernel_launch(void* const* d_in, const int* in_sizes, int n_in,
                              void* d_out, int out_size, void* d_ws, size_t ws_size,
                              hipStream_t stream) {
  const float* emb_table = (const float*)d_in[0];
  const float* W1        = (const float*)d_in[1];
  const float* b1        = (const float*)d_in[2];
  const float* W2        = (const float*)d_in[3];
  const float* b2        = (const float*)d_in[4];
  const int*   ids       = (const int*)d_in[5];
  const int*   lengths   = (const int*)d_in[6];
  float*       out       = (float*)d_out;

  const size_t shmem = sizeof(Smem);   // ~90 KB -> 3 workgroups per WGP (320 KB LDS)
  hipFuncSetAttribute(reinterpret_cast<const void*>(attn_pool_kernel),
                      hipFuncAttributeMaxDynamicSharedMemorySize, (int)shmem);
  attn_pool_kernel<<<B_, 256, shmem, stream>>>(emb_table, W1, b1, W2, b2, ids, lengths, out);
}
